// UNetTransformerWithAttentionFusion_41764261987153
// MI455X (gfx1250) — matrix operand order repaired and checked by reading, not compiled, not run on hardware
//
#include <hip/hip_runtime.h>
#include <stdint.h>

// ---------------------------------------------------------------------------
// MI455X (gfx1250) fused sparse-attention block.
// All GEMMs run on v_wmma_f32_16x16x32_bf16 (fp32 accumulate).
// Attention fused per 16-query tile: the 256 MB score tensor stays in LDS.
// V is stored transposed ([bh][hd][seq]) so every WMMA B-fragment is two
// contiguous 16-byte loads.
// ---------------------------------------------------------------------------

typedef __bf16 bf16;
typedef __attribute__((ext_vector_type(16))) __bf16 v16bf;
typedef __attribute__((ext_vector_type(8)))  float  v8f;
typedef __attribute__((ext_vector_type(4)))  unsigned int v4u;

#define WMMA_BF16(A, Bm, Cm) \
  __builtin_amdgcn_wmma_f32_16x16x32_bf16(false, (A), false, (Bm), (short)0, (Cm), false, false)

static constexpr int BATCH = 8;
static constexpr int SEQ   = 1024;
static constexpr int CH    = 512;
static constexpr int HEADS = 8;
static constexpr int HD    = 64;      // head dim
static constexpr int TOPK  = 128;
static constexpr int PAD   = 40;      // LDS row pitch (elems): conflict-free frags

// native f32 -> bf16 (v_cvt_pk_bf16_f32 on gfx1250)
__device__ __forceinline__ bf16 f2bf(float f) { return (bf16)f; }

// order-preserving float <-> uint key (larger float -> larger key)
__device__ __forceinline__ unsigned f2key(float f) {
  union { float f; unsigned u; } c; c.f = f;
  return (c.u & 0x80000000u) ? ~c.u : (c.u | 0x80000000u);
}
__device__ __forceinline__ float key2f(unsigned k) {
  unsigned bits = (k & 0x80000000u) ? (k ^ 0x80000000u) : ~k;
  union { unsigned u; float f; } c; c.u = bits;
  return c.f;
}

// wave32 butterfly reductions via DS lane swizzle (xor-mask mode)
__device__ __forceinline__ int wave_isum(int v) {
  v += __builtin_amdgcn_ds_swizzle(v, 0x041F);
  v += __builtin_amdgcn_ds_swizzle(v, 0x081F);
  v += __builtin_amdgcn_ds_swizzle(v, 0x101F);
  v += __builtin_amdgcn_ds_swizzle(v, 0x201F);
  v += __builtin_amdgcn_ds_swizzle(v, 0x401F);
  return v;
}
__device__ __forceinline__ float wave_fsum(float v) {
  v += __int_as_float(__builtin_amdgcn_ds_swizzle(__float_as_int(v), 0x041F));
  v += __int_as_float(__builtin_amdgcn_ds_swizzle(__float_as_int(v), 0x081F));
  v += __int_as_float(__builtin_amdgcn_ds_swizzle(__float_as_int(v), 0x101F));
  v += __int_as_float(__builtin_amdgcn_ds_swizzle(__float_as_int(v), 0x201F));
  v += __int_as_float(__builtin_amdgcn_ds_swizzle(__float_as_int(v), 0x401F));
  return v;
}

// Load a v16bf fragment from two 16-byte runs.
union V16U { v16bf v; v4u q[2]; };
__device__ __forceinline__ v16bf ld_frag2(const bf16* p0, const bf16* p1) {
  V16U t;
  t.q[0] = *reinterpret_cast<const v4u*>(p0);
  t.q[1] = *reinterpret_cast<const v4u*>(p1);
  return t.v;
}

union PK4 { bf16 h[4]; unsigned long long u; };
union PK8 { bf16 h[8]; v4u u; };

// ===========================================================================
// Kernel 1: fused Q/K/V projection.  Y = x@[Wq|Wkv] + [bq|bkv]  (1536 cols).
// q,k -> [bh][seq][hd] bf16;  V -> TRANSPOSED [bh][hd][seq] bf16.
// Block tile 128x128, 8 waves of 64x32, K = 512 in 32-chunks.
// Each 128-col block lies entirely in one section: 0-3 q, 4-7 k, 8-11 v.
// ===========================================================================
__global__ __launch_bounds__(256) void qkv_kernel(
    const float* __restrict__ x,  const float* __restrict__ Wq,
    const float* __restrict__ bq, const float* __restrict__ Wkv,
    const float* __restrict__ bkv,
    bf16* __restrict__ qw, bf16* __restrict__ kw, bf16* __restrict__ vtw)
{
  __shared__ bf16 Abuf[128][PAD];        // x tile, row-major
  __shared__ bf16 BbufT[128][PAD];       // W tile, transposed: [n][k]
  __shared__ bf16 Obuf[8][64][PAD];      // per-wave epilogue transpose tile

  const int t = threadIdx.x, lane = t & 31, wave = t >> 5;
  const int wm = wave >> 2, wn = wave & 3;           // 2x4 wave grid
  const int row0 = blockIdx.y * 128;
  const int col0 = blockIdx.x * 128;
  const int sect = col0 >> 9;                        // 0=q, 1=k, 2=v

  v8f acc[4][2];
  const v8f vzero = {0.f,0.f,0.f,0.f,0.f,0.f,0.f,0.f};
#pragma unroll
  for (int i = 0; i < 4; ++i) { acc[i][0] = vzero; acc[i][1] = vzero; }

  for (int k0 = 0; k0 < CH; k0 += 32) {
    // -- gather phase: wide coalesced global loads into registers --
    float4 av[4];
#pragma unroll
    for (int i = 0; i < 4; ++i) {                    // A: float4 per thread
      int f = t + i * 256, r = f >> 3, c4 = (f & 7) * 4;
      av[i] = *reinterpret_cast<const float4*>(&x[(size_t)(row0 + r) * CH + k0 + c4]);
    }
    float wv[4][4];
#pragma unroll
    for (int i = 0; i < 4; ++i) {                    // B: 4 consecutive k @ fixed n
      int f = t + i * 256, n = f & 127, kk4 = (f >> 7) * 4;
      int c = col0 + n;
      const float* wp;
      size_t ldw;
      if (c < CH)      { wp = &Wq[(size_t)(k0 + kk4) * CH + c];              ldw = CH; }
      else             { wp = &Wkv[(size_t)(k0 + kk4) * (2*CH) + (c - CH)];  ldw = 2*CH; }
#pragma unroll
      for (int j = 0; j < 4; ++j) wv[i][j] = wp[(size_t)j * ldw];
    }
    // -- scatter phase: packed bf16 LDS stores --
#pragma unroll
    for (int i = 0; i < 4; ++i) {
      int f = t + i * 256, r = f >> 3, c4 = (f & 7) * 4;
      PK4 pk;
      pk.h[0] = f2bf(av[i].x); pk.h[1] = f2bf(av[i].y);
      pk.h[2] = f2bf(av[i].z); pk.h[3] = f2bf(av[i].w);
      *reinterpret_cast<unsigned long long*>(&Abuf[r][c4]) = pk.u;
    }
#pragma unroll
    for (int i = 0; i < 4; ++i) {
      int f = t + i * 256, n = f & 127, kk4 = (f >> 7) * 4;
      PK4 pk;
#pragma unroll
      for (int j = 0; j < 4; ++j) pk.h[j] = f2bf(wv[i][j]);
      *reinterpret_cast<unsigned long long*>(&BbufT[n][kk4]) = pk.u;
    }
    __syncthreads();

    v16bf bfr[2];
#pragma unroll
    for (int j = 0; j < 2; ++j) {
      const bf16* p = &BbufT[wn * 32 + j * 16 + (lane & 15)][(lane < 16) ? 0 : 8];
      bfr[j] = ld_frag2(p, p + 16);
    }
#pragma unroll
    for (int i = 0; i < 4; ++i) {
      const bf16* p0 = &Abuf[wm * 64 + i * 16 + (lane & 15)][(lane < 16) ? 0 : 8];
      v16bf af = ld_frag2(p0, p0 + 16);
      acc[i][0] = WMMA_BF16(af, bfr[0], acc[i][0]);
      acc[i][1] = WMMA_BF16(af, bfr[1], acc[i][1]);
    }
    __syncthreads();
  }

  const int hi = (lane >> 4) << 3;
  if (sect < 2) {
    // ---- Q/K epilogue: per-wave LDS transpose, then b128 runs along hd ----
#pragma unroll
    for (int i = 0; i < 4; ++i)
#pragma unroll
      for (int j = 0; j < 2; ++j) {
        int c = col0 + wn * 32 + j * 16 + (lane & 15);
        float bias = (c < CH) ? bq[c] : bkv[c - CH];
#pragma unroll
        for (int r = 0; r < 8; ++r)
          Obuf[wave][i * 16 + r + hi][j * 16 + (lane & 15)] = f2bf(acc[i][j][r] + bias);
      }
    __syncthreads();
    bf16* dst = (sect == 0) ? qw : kw;
    const int c0  = col0 + wn * 32;
    const int dd0 = c0 & 63;
    const int h   = (c0 & 511) >> 6;
#pragma unroll
    for (int rr = 0; rr < 2; ++rr) {
      int r = lane * 2 + rr;                         // 0..63 within wave tile
      int m = row0 + wm * 64 + r;
      int b = m >> 10, nn = m & 1023;
      bf16* gp = dst + (((size_t)(b * HEADS + h)) * SEQ + nn) * HD + dd0;
#pragma unroll
      for (int qd = 0; qd < 4; ++qd)
        *reinterpret_cast<v4u*>(gp + qd * 8) =
            *reinterpret_cast<const v4u*>(&Obuf[wave][r][qd * 8]);
    }
  } else {
    // ---- V epilogue: rows are consecutive seq -> direct b128 stores ----
#pragma unroll
    for (int i = 0; i < 4; ++i)
#pragma unroll
      for (int j = 0; j < 2; ++j) {
        int c  = col0 + wn * 32 + j * 16 + (lane & 15);
        int c2 = c - 2 * CH, h = c2 >> 6, dd = c2 & 63;
        float bias = bkv[c - CH];
        int n0 = row0 + wm * 64 + i * 16 + hi;
        int b = n0 >> 10, nn = n0 & 1023;
        PK8 pk;
#pragma unroll
        for (int r = 0; r < 8; ++r) pk.h[r] = f2bf(acc[i][j][r] + bias);
        *reinterpret_cast<v4u*>(
            &vtw[(((size_t)(b * HEADS + h)) * HD + dd) * SEQ + nn]) = pk.u;
      }
  }
}

// ===========================================================================
// Kernel 2: fused sparse attention for one (b,h) and one 16-query tile.
//  phase 1: S = (Q Kᵀ)*scale -> LDS                      (WMMA)
//  phase 2: per-row exact 128-th-largest threshold via 32-step MSB binary
//           search on order keys (ds_swizzle wave reductions);
//           weights w = topk ? exp(s) : 1  (faithful softmax(attn*mask))
//  phase 3: O = W · Vᵀ (WMMA, K split across wave pairs), /denom -> ctx bf16
// ===========================================================================
__global__ __launch_bounds__(256) void attn_kernel(
    const bf16* __restrict__ qw, const bf16* __restrict__ kw,
    const bf16* __restrict__ vtw, bf16* __restrict__ ctx)
{
  extern __shared__ char smem[];
  float* scores = (float*)smem;                                  // [16][1024] 64KB
  bf16*  wmat   = (bf16*)(smem + 16 * SEQ * 4);                  // [16][1024] 32KB
  float* opart  = (float*)(smem + 16 * SEQ * 4 + 16 * SEQ * 2);  // [2][16][64] 8KB
  float* denom  = (float*)(smem + 16 * SEQ * 4 + 16 * SEQ * 2 + 2 * 16 * HD * 4);

  const int t = threadIdx.x, lane = t & 31, wave = t >> 5;
  const int rowTile = blockIdx.x;                 // 0..63
  const int bh      = blockIdx.y;                 // 0..63
  const size_t qbase  = ((size_t)bh * SEQ + rowTile * 16) * HD;
  const size_t kvbase = (size_t)bh * SEQ * HD;    // same extent for K and Vᵀ
  const v8f vzero = {0.f,0.f,0.f,0.f,0.f,0.f,0.f,0.f};

  // ---- phase 1: scores ----
  v16bf qf[2];
  {
    const bf16* qrow = qw + qbase + (size_t)(lane & 15) * HD;
#pragma unroll
    for (int j = 0; j < 2; ++j) {
      const bf16* p0 = qrow + j * 32 + ((lane < 16) ? 0 : 8);
      qf[j] = ld_frag2(p0, p0 + 16);
    }
  }
  const int hi = (lane >> 4) << 3;
  for (int it = 0; it < 8; ++it) {
    int mk = (wave + it * 8) * 16;                 // key tile base
    v8f acc = vzero;
#pragma unroll
    for (int j = 0; j < 2; ++j) {                  // K = 64 = 2 x 32
      const bf16* p = kw + kvbase + (size_t)(mk + (lane & 15)) * HD
                    + j * 32 + ((lane < 16) ? 0 : 8);
      v16bf bk = ld_frag2(p, p + 16);
      acc = WMMA_BF16(qf[j], bk, acc);
    }
#pragma unroll
    for (int r = 0; r < 8; ++r)
      scores[(r + hi) * SEQ + mk + (lane & 15)] = acc[r] * 0.125f;  // d^-0.5
  }
  __syncthreads();

  // ---- phase 2: threshold + weights (2 rows per wave) ----
  for (int rr = 0; rr < 2; ++rr) {
    int row = wave * 2 + rr;
    unsigned keys[32];
#pragma unroll
    for (int i = 0; i < 32; ++i)
      keys[i] = f2key(scores[row * SEQ + i * 32 + lane]);

    unsigned thr = 0u;
    for (int bit = 31; bit >= 0; --bit) {          // largest T with cnt(>=T)>=128
      unsigned cand = thr | (1u << bit);
      int c = 0;
#pragma unroll
      for (int i = 0; i < 32; ++i) c += (keys[i] >= cand);
      c = wave_isum(c);
      if (c >= TOPK) thr = cand;
    }
    float dsum = 0.f;
#pragma unroll
    for (int i = 0; i < 32; ++i) {
      float s = key2f(keys[i]);
      float w = (keys[i] >= thr) ? __expf(s) : 1.0f;   // masked -> exp(0)=1
      dsum += w;
      wmat[row * SEQ + i * 32 + lane] = f2bf(w);
    }
    dsum = wave_fsum(dsum);
    if (lane == 0) denom[row] = dsum;
  }
  __syncthreads();

  // ---- phase 3: O = W x Vᵀ (contiguous B-fragments from transposed V) ----
  const int tile = wave & 3;                       // output 16-col tile of d=64
  const int hh   = wave >> 2;                      // K half: 0..511 / 512..1023
  v8f acc = vzero;
  const bf16* vcol = vtw + kvbase + (size_t)(tile * 16 + (lane & 15)) * SEQ;
  for (int kc = 0; kc < 16; ++kc) {
    int k0 = hh * 512 + kc * 32;
    const bf16* p0 = &wmat[(lane & 15) * SEQ + k0 + ((lane < 16) ? 0 : 8)];
    v16bf af = ld_frag2(p0, p0 + 16);
    const bf16* pb = vcol + k0 + ((lane < 16) ? 0 : 8);
    v16bf bv = ld_frag2(pb, pb + 16);
    acc = WMMA_BF16(af, bv, acc);
  }
#pragma unroll
  for (int r = 0; r < 8; ++r)
    opart[(hh * 16 + r + hi) * HD + tile * 16 + (lane & 15)] = acc[r];
  __syncthreads();

  // combine K-halves, normalize, write ctx in [B,N,C] bf16
  const int b = bh >> 3, h = bh & 7;
#pragma unroll
  for (int i = 0; i < 4; ++i) {
    int e = t + i * 256, m = e >> 6, c = e & 63;
    float o = (opart[m * HD + c] + opart[(16 + m) * HD + c]) / denom[m];
    int nrow = rowTile * 16 + m;
    ctx[((size_t)(b * SEQ + nrow)) * CH + h * HD + c] = f2bf(o);
  }
}

// ===========================================================================
// Kernel 3: out = ctx @ Wp + bp  (fp32 output). Same tiling as kernel 1.
// ===========================================================================
__global__ __launch_bounds__(256) void proj_kernel(
    const bf16* __restrict__ ctx, const float* __restrict__ Wp,
    const float* __restrict__ bp, float* __restrict__ out)
{
  __shared__ bf16 Abuf[128][PAD];
  __shared__ bf16 BbufT[128][PAD];

  const int t = threadIdx.x, lane = t & 31, wave = t >> 5;
  const int wm = wave >> 2, wn = wave & 3;
  const int row0 = blockIdx.y * 128;               // 64 blocks
  const int col0 = blockIdx.x * 128;               // 4 blocks (512)

  v8f acc[4][2];
  const v8f vzero = {0.f,0.f,0.f,0.f,0.f,0.f,0.f,0.f};
#pragma unroll
  for (int i = 0; i < 4; ++i) { acc[i][0] = vzero; acc[i][1] = vzero; }

  for (int k0 = 0; k0 < CH; k0 += 32) {
    // gather
    v4u cv[2];
#pragma unroll
    for (int i = 0; i < 2; ++i) {                  // ctx already bf16: b128 loads
      int f = t + i * 256, r = f >> 2, c8 = (f & 3) * 8;
      cv[i] = *reinterpret_cast<const v4u*>(&ctx[(size_t)(row0 + r) * CH + k0 + c8]);
    }
    float wv[4][4];
#pragma unroll
    for (int i = 0; i < 4; ++i) {
      int f = t + i * 256, n = f & 127, kk4 = (f >> 7) * 4;
#pragma unroll
      for (int j = 0; j < 4; ++j)
        wv[i][j] = Wp[(size_t)(k0 + kk4 + j) * CH + col0 + n];
    }
    // scatter
#pragma unroll
    for (int i = 0; i < 2; ++i) {
      int f = t + i * 256, r = f >> 2, c8 = (f & 3) * 8;
      *reinterpret_cast<v4u*>(&Abuf[r][c8]) = cv[i];
    }
#pragma unroll
    for (int i = 0; i < 4; ++i) {
      int f = t + i * 256, n = f & 127, kk4 = (f >> 7) * 4;
      PK4 pk;
#pragma unroll
      for (int j = 0; j < 4; ++j) pk.h[j] = f2bf(wv[i][j]);
      *reinterpret_cast<unsigned long long*>(&BbufT[n][kk4]) = pk.u;
    }
    __syncthreads();

    v16bf bfr[2];
#pragma unroll
    for (int j = 0; j < 2; ++j) {
      const bf16* p = &BbufT[wn * 32 + j * 16 + (lane & 15)][(lane < 16) ? 0 : 8];
      bfr[j] = ld_frag2(p, p + 16);
    }
#pragma unroll
    for (int i = 0; i < 4; ++i) {
      const bf16* p0 = &Abuf[wm * 64 + i * 16 + (lane & 15)][(lane < 16) ? 0 : 8];
      v16bf af = ld_frag2(p0, p0 + 16);
      acc[i][0] = WMMA_BF16(af, bfr[0], acc[i][0]);
      acc[i][1] = WMMA_BF16(af, bfr[1], acc[i][1]);
    }
    __syncthreads();
  }

  const int hi = (lane >> 4) << 3;
#pragma unroll
  for (int i = 0; i < 4; ++i)
#pragma unroll
    for (int j = 0; j < 2; ++j) {
      int c = col0 + wn * 32 + j * 16 + (lane & 15);
      float bias = bp[c];
#pragma unroll
      for (int r = 0; r < 8; ++r) {
        int m = row0 + wm * 64 + i * 16 + r + hi;
        out[(size_t)m * CH + c] = acc[i][j][r] + bias;   // fp32, 128B/instr
      }
    }
}

// ===========================================================================
extern "C" void kernel_launch(void* const* d_in, const int* in_sizes, int n_in,
                              void* d_out, int out_size, void* d_ws, size_t ws_size,
                              hipStream_t stream) {
  const float* x   = (const float*)d_in[0];
  const float* Wq  = (const float*)d_in[1];
  const float* bq  = (const float*)d_in[2];
  const float* Wkv = (const float*)d_in[3];
  const float* bkv = (const float*)d_in[4];
  const float* Wp  = (const float*)d_in[5];
  const float* bp  = (const float*)d_in[6];
  float* out = (float*)d_out;

  // workspace: q/k in [bh][seq][hd], V transposed [bh][hd][seq] (8 MB each),
  // ctx in [B,N,C] bf16 (8 MB)  -> 32 MB total
  char* ws = (char*)d_ws;
  bf16* qw   = (bf16*)(ws);
  bf16* kw   = (bf16*)(ws + (size_t)8  * 1024 * 1024);
  bf16* vtw  = (bf16*)(ws + (size_t)16 * 1024 * 1024);
  bf16* ctxb = (bf16*)(ws + (size_t)24 * 1024 * 1024);

  qkv_kernel<<<dim3(12, 64), 256, 0, stream>>>(x, Wq, bq, Wkv, bkv, qw, kw, vtw);

  constexpr size_t ATTN_SMEM = (size_t)16 * SEQ * 4      // scores
                             + (size_t)16 * SEQ * 2      // weights (bf16)
                             + (size_t)2 * 16 * HD * 4   // K-split partials
                             + 16 * 4;                   // denominators
  attn_kernel<<<dim3(64, 64), 256, ATTN_SMEM, stream>>>(qw, kw, vtw, ctxb);

  proj_kernel<<<dim3(4, 64), 256, 0, stream>>>(ctxb, Wp, bp, out);
}
